// CoreEncoder_48696339202072
// MI455X (gfx1250) — matrix operand order repaired and checked by reading, not compiled, not run on hardware
//
#include <hip/hip_runtime.h>

// ---------------------------------------------------------------------------
// CDNA5 (gfx1250, wave32) fused GRU/conv encoder.
//  - bf16 WMMA 16x16x32, f32 accumulation
//  - phase 0: repack all weights into wave32 B-operand swizzle (bf16)
//  - phase 1: streaming recurrent kernel, 8 WGs (16 batch rows each), T=512
//  - phase 2: WMMA GEMM heads (z, s1+tanh, s2+tanh) over 65536 rows
//  - branch-free transcendentals (v_tanh_f32 if available, else v_exp/v_rcp)
// ---------------------------------------------------------------------------

typedef __attribute__((ext_vector_type(16))) __bf16 v16bf;
typedef __attribute__((ext_vector_type(8)))  float  v8f;

#define NB     128
#define TSTEPS 512
#define CH     544
#define LDSP   560   // LDS row pitch (bf16 elements); 70 uint4 per row

__device__ __forceinline__ unsigned short f2bfb(float f) {
  unsigned u = __float_as_uint(f);
  return (unsigned short)((u + 0x7FFFu + ((u >> 16) & 1u)) >> 16);  // RNE
}
__device__ __forceinline__ __bf16 f2bf(float f) {
  union { unsigned short u; __bf16 b; } x; x.u = f2bfb(f); return x.b;
}
__device__ __forceinline__ float bf2f(__bf16 b) {
  union { unsigned short u; __bf16 b; } x; x.b = b;
  return __uint_as_float(((unsigned)x.u) << 16);
}

// Branch-free transcendentals (avoid OCML's EXEC-diverging tanhf).
__device__ __forceinline__ float fast_tanh(float x) {
#if __has_builtin(__builtin_amdgcn_tanhf)
  return __builtin_amdgcn_tanhf(x);          // native v_tanh_f32 (gfx1250 TRANS)
#else
  float ax = fabsf(x);
  float e  = __builtin_amdgcn_exp2f(ax * -2.8853900817779268f);   // exp(-2|x|) in (0,1]
  float t  = (1.f - e) * __builtin_amdgcn_rcpf(1.f + e);          // tanh(|x|)
  unsigned s = (__float_as_uint(x) & 0x80000000u) | __float_as_uint(t);
  return __uint_as_float(s);
#endif
}
__device__ __forceinline__ float fast_sigmoid(float x) {
  float e = __builtin_amdgcn_exp2f(x * -1.4426950408889634f);     // exp(-x)
  return __builtin_amdgcn_rcpf(1.f + e);
}

__device__ __forceinline__ v8f wmma_bf(v16bf a, v16bf b, v8f c) {
  // v_wmma_f32_16x16x32_bf16 : D = A(16x32) x B(32x16) + C
  return __builtin_amdgcn_wmma_f32_16x16x32_bf16(false, a, false, b, (short)0, c, false, false);
}

// B operand: pre-swizzled, 512 bf16 per 16(N)x32(K) tile, lane-major 16 each.
__device__ __forceinline__ v16bf loadB(const __bf16* __restrict__ p) {
  return *(const v16bf*)p;  // 32B contiguous per lane
}

// A operand gather from row-major bf16 [16 rows x K] with arbitrary row pitch.
// ISA 16-bit A layout: lanes0-15 M=lane; VGPR j<4 -> K=grp*8+2j, j>=4 -> K=16+grp*8+2(j-4)
__device__ __forceinline__ v16bf gatherA_bf(const __bf16* base, long pitch, int k0, int lane) {
  int m = lane & 15, grp = lane >> 4;
  const __bf16* row = base + (long)m * pitch + k0 + grp * 8;
  v16bf a; unsigned* au = (unsigned*)&a;
#pragma unroll
  for (int j = 0; j < 8; ++j) {
    int kk = (j < 4) ? (2 * j) : (16 + 2 * (j - 4));
    au[j] = *(const unsigned*)(row + kk);
  }
  return a;
}

// ---------------------------------------------------------------------------
// Phase 0: weight repack f32 -> bf16 B-operand swizzle.
// dst[(nt*KT+kt)*512 + L*16 + e] = W[n= nt*16+L%16][k= kt*32+(L/16)*8+(e%8)+(e/8)*16]
// conv_cin>0: combined K = [past(cin) | cur(cin)], src layout (32,cin,2).
// ---------------------------------------------------------------------------
__global__ void repack_kernel(const float* __restrict__ W, __bf16* __restrict__ dst,
                              int NT, int KT, int srcN, int srcK, int conv_cin) {
  int total = NT * KT * 512;
  for (int idx = blockIdx.x * blockDim.x + threadIdx.x; idx < total;
       idx += gridDim.x * blockDim.x) {
    int tile = idx >> 9, rem = idx & 511;
    int L = rem >> 4, e = rem & 15;
    int nt = tile / KT, kt = tile - nt * KT;
    int n = nt * 16 + (L & 15);
    int k = kt * 32 + (L >> 4) * 8 + (e & 7) + (e >> 3) * 16;
    float v = 0.f;
    if (n < srcN && k < srcK) {
      if (conv_cin > 0) {
        int tap = (k < conv_cin) ? 0 : 1;
        int c   = (k < conv_cin) ? k : k - conv_cin;
        v = W[(n * conv_cin + c) * 2 + tap];
      } else {
        v = W[n * srcK + k];
      }
    }
    dst[idx] = f2bf(v);
  }
}

// ---------------------------------------------------------------------------
// Phase 1: streaming recurrent kernel. grid=8 (batch tiles of 16), block=256.
// ---------------------------------------------------------------------------
struct StreamParams {
  const float* feat;                       // [128][1024][20]
  const __bf16* d1p;  const float* d1b;
  const __bf16* wihp[5]; const float* bihp[5];
  const __bf16* whhp[5]; const float* bhhp[5];
  const __bf16* convp[5]; const float* convb[5];
  __bf16* xall;                            // [128][512][544] bf16
};

// LDS bias cache layout (floats):
//   [0,64)            d1b
//   [64 + l*192)      bih[l]
//   [1024 + l*192)    bhh[l]
//   [1984 + l*32)     convb[l]
#define BSH_TOTAL 2144

__global__ __launch_bounds__(256) void stream_kernel(StreamParams p) {
  __shared__ __bf16 xc[16 * LDSP];         // x[t] (current), bf16
  __shared__ __bf16 xh[2][16 * LDSP];      // ring: x[t-1], x[t-2]
  __shared__ __bf16 fstage[16 * 64];       // staged feature frame (40 cols + zero pad)
  __shared__ float  xps[12 * 256];         // staged input-projection tiles (f32)
  __shared__ float  cpart[8 * 256];        // conv partial accumulators
  __shared__ float  bsh[BSH_TOTAL];        // bias cache

  const int tid  = threadIdx.x;
  const int w    = tid >> 5;               // wave id 0..7
  const int lane = tid & 31;
  const int m    = lane & 15;
  const int grp  = lane >> 4;
  const int bt   = blockIdx.x;             // batch tile 0..7

  // one-time init: zero history ring + fstage pad; cache biases
  {
    uint4 z4; z4.x = z4.y = z4.z = z4.w = 0u;
    uint4* xh4 = (uint4*)xh;
    for (int i = tid; i < 2 * 16 * (LDSP / 8); i += 256) xh4[i] = z4;
    for (int i = tid; i < 16 * 64; i += 256) fstage[i] = f2bf(0.f);
    if (tid < 64) bsh[tid] = p.d1b[tid];
#pragma unroll
    for (int l = 0; l < 5; ++l) {
      for (int i = tid; i < 192; i += 256) {
        bsh[64 + l * 192 + i]   = p.bihp[l][i];
        bsh[1024 + l * 192 + i] = p.bhhp[l][i];
      }
      if (tid < 32) bsh[1984 + l * 32 + tid] = p.convb[l][tid];
    }
  }
  __syncthreads();

  const int IN_[5]  = {64, 160, 256, 352, 448};   // GRU input widths
  const int CIN_[5] = {128, 224, 320, 416, 512};  // conv input widths
  const int KTI_[5] = {2, 5, 8, 11, 14};          // wih K-tiles

#pragma unroll 1
  for (int t = 0; t < TSTEPS; ++t) {
    // ---- stage feature frame pair into LDS (16 rows x 40 f32 -> bf16) ----
    for (int i = tid; i < 16 * 40; i += 256) {
      int row = i / 40, c = i - row * 40;
      float v = p.feat[(long)(bt * 16 + row) * 20480 + t * 40 + c];
      fstage[row * 64 + c] = f2bf(v);
    }
    __syncthreads();

    // ---------------- d1: x[0:64) = tanh(feat2 @ W1^T + b) --------------
    if (w < 4) {
      v8f acc = {};
#pragma unroll
      for (int kt = 0; kt < 2; ++kt) {
        v16bf a = gatherA_bf(fstage, 64, kt * 32, lane);
        v16bf b = loadB(p.d1p + (w * 2 + kt) * 512 + lane * 16);
        acc = wmma_bf(a, b, acc);
      }
      float bias = bsh[w * 16 + m];
#pragma unroll
      for (int i = 0; i < 8; ++i) {
        int row = i + 8 * grp;
        xc[row * LDSP + w * 16 + m] = f2bf(fast_tanh(acc[i] + bias));
      }
    }
    __syncthreads();

    const __bf16* hprev = &xh[(t + 1) & 1][0];   // slot holding x[t-1] (zeros at t=0)

#pragma unroll
    for (int l = 0; l < 5; ++l) {
      const int IN = IN_[l], CIN = CIN_[l], KT = KTI_[l];

      // ---- GRU input projection: xp = x[0:IN) @ wih^T + bih (N=192) ----
      for (int nt = w; nt < 12; nt += 8) {
        v8f acc = {};
        for (int kt = 0; kt < KT; ++kt) {
          v16bf a = gatherA_bf(xc, LDSP, kt * 32, lane);
          v16bf b = loadB(p.wihp[l] + (nt * KT + kt) * 512 + lane * 16);
          acc = wmma_bf(a, b, acc);
        }
        float bias = bsh[64 + l * 192 + nt * 16 + m];
#pragma unroll
        for (int i = 0; i < 8; ++i) xps[nt * 256 + lane * 8 + i] = acc[i] + bias;
      }
      __syncthreads();

      // ---- GRU gates: waves 0-3, gate column w (channels w*16..w*16+15) ----
      if (w < 4) {
        v8f gr = {}, gz = {}, gn = {};
#pragma unroll
        for (int kt = 0; kt < 2; ++kt) {
          v16bf a = gatherA_bf(hprev, LDSP, IN + kt * 32, lane);
          gr = wmma_bf(a, loadB(p.whhp[l] + ((w      * 2) + kt) * 512 + lane * 16), gr);
          gz = wmma_bf(a, loadB(p.whhp[l] + (((w + 4) * 2) + kt) * 512 + lane * 16), gz);
          gn = wmma_bf(a, loadB(p.whhp[l] + (((w + 8) * 2) + kt) * 512 + lane * 16), gn);
        }
        float br = bsh[1024 + l * 192 + w * 16 + m];
        float bz = bsh[1024 + l * 192 + 64 + w * 16 + m];
        float bn = bsh[1024 + l * 192 + 128 + w * 16 + m];
#pragma unroll
        for (int i = 0; i < 8; ++i) {
          float xr = xps[w * 256 + lane * 8 + i];
          float xz = xps[(w + 4) * 256 + lane * 8 + i];
          float xn = xps[(w + 8) * 256 + lane * 8 + i];
          float r = fast_sigmoid(xr + gr[i] + br);
          float z = fast_sigmoid(xz + gz[i] + bz);
          float n = fast_tanh(xn + r * (gn[i] + bn));
          int row = i + 8 * grp;
          float hp = bf2f(hprev[row * LDSP + IN + w * 16 + m]);
          float h = (1.f - z) * n + z * hp;
          xc[row * LDSP + IN + w * 16 + m] = f2bf(h);
        }
      }
      __syncthreads();

      // ---- conv: out = tanh(W0@past + W1@cur + b), K=2*CIN split 4-way ----
      {
        const int d = (l == 0) ? 1 : 2;
        int pt = t - d; if (pt < 0) pt = 0;
        const __bf16* past = (pt == t) ? xc : &xh[pt & 1][0];
        const int KT2 = CIN / 16;            // K tiles over 2*CIN
        const int nt = w & 1, q = w >> 1, per = KT2 >> 2;
        v8f acc = {};
        for (int kk = q * per; kk < (q + 1) * per; ++kk) {
          int kg = kk * 32;
          const __bf16* src = (kg < CIN) ? past : xc;
          int k0 = (kg < CIN) ? kg : kg - CIN;
          v16bf a = gatherA_bf(src, LDSP, k0, lane);
          v16bf b = loadB(p.convp[l] + (nt * KT2 + kk) * 512 + lane * 16);
          acc = wmma_bf(a, b, acc);
        }
#pragma unroll
        for (int i = 0; i < 8; ++i) cpart[w * 256 + lane * 8 + i] = acc[i];
        __syncthreads();
        if (w < 2) {
          float bias = bsh[1984 + l * 32 + w * 16 + m];
#pragma unroll
          for (int i = 0; i < 8; ++i) {
            float s = cpart[w * 256 + lane * 8 + i] + cpart[(w + 2) * 256 + lane * 8 + i]
                    + cpart[(w + 4) * 256 + lane * 8 + i] + cpart[(w + 6) * 256 + lane * 8 + i]
                    + bias;
            int row = i + 8 * grp;
            xc[row * LDSP + CIN + w * 16 + m] = f2bf(fast_tanh(s));
          }
        }
        __syncthreads();
      }
    }

    // ---- commit: xc -> ring slot (t&1) and to global xall (uint4) ----
    {
      uint4* dh = (uint4*)&xh[t & 1][0];
      const uint4* sc = (const uint4*)xc;       // 70 uint4 per LDS row
      for (int i = tid; i < 16 * 68; i += 256) {  // 68 uint4 = 544 bf16
        int row = i / 68, c = i - row * 68;
        uint4 v = sc[row * 70 + c];
        dh[row * 70 + c] = v;
        *((uint4*)(p.xall + ((long)(bt * 16 + row) * TSTEPS + t) * CH) + c) = v;
      }
    }
    __syncthreads();
  }
}

// ---------------------------------------------------------------------------
// Phase 2: output heads (parallel WMMA GEMMs over all 65536 rows).
// ---------------------------------------------------------------------------
__global__ __launch_bounds__(256) void head_z(const __bf16* __restrict__ xall,
                                              const __bf16* __restrict__ wp,
                                              const float* __restrict__ bias,
                                              float* __restrict__ out) {
  int job = blockIdx.x * 8 + (threadIdx.x >> 5);   // 8*512*5 jobs
  int lane = threadIdx.x & 31, m = lane & 15, grp = lane >> 4;
  int nt = job % 5; int r1 = job / 5; int t = r1 & 511; int bt = r1 >> 9;
  const __bf16* base = xall + (long)bt * 16 * TSTEPS * CH + (long)t * CH;
  v8f acc = {};
  for (int kt = 0; kt < 17; ++kt) {
    v16bf a = gatherA_bf(base, (long)TSTEPS * CH, kt * 32, lane);
    v16bf b = loadB(wp + (nt * 17 + kt) * 512 + lane * 16);
    acc = wmma_bf(a, b, acc);
  }
  float bv = bias[nt * 16 + m];
#pragma unroll
  for (int i = 0; i < 8; ++i) {
    int row = bt * 16 + i + 8 * grp;
    out[((long)row * TSTEPS + t) * 80 + nt * 16 + m] = acc[i] + bv;
  }
}

__global__ __launch_bounds__(256) void head_s1(const __bf16* __restrict__ xall,
                                               const __bf16* __restrict__ wp,
                                               const float* __restrict__ bias,
                                               __bf16* __restrict__ s1buf) {
  int job = blockIdx.x * 8 + (threadIdx.x >> 5);   // 8*512*8 jobs
  int lane = threadIdx.x & 31, m = lane & 15, grp = lane >> 4;
  int nt = job & 7; int r1 = job >> 3; int t = r1 & 511; int bt = r1 >> 9;
  const __bf16* base = xall + (long)bt * 16 * TSTEPS * CH + (long)t * CH;
  v8f acc = {};
  for (int kt = 0; kt < 17; ++kt) {
    v16bf a = gatherA_bf(base, (long)TSTEPS * CH, kt * 32, lane);
    v16bf b = loadB(wp + (nt * 17 + kt) * 512 + lane * 16);
    acc = wmma_bf(a, b, acc);
  }
  float bv = bias[nt * 16 + m];
#pragma unroll
  for (int i = 0; i < 8; ++i) {
    int row = bt * 16 + i + 8 * grp;
    s1buf[((long)row * TSTEPS + t) * 128 + nt * 16 + m] = f2bf(fast_tanh(acc[i] + bv));
  }
}

__global__ __launch_bounds__(256) void head_s2(const __bf16* __restrict__ s1buf,
                                               const __bf16* __restrict__ wp,
                                               const float* __restrict__ bias,
                                               float* __restrict__ out) {
  int job = blockIdx.x * 8 + (threadIdx.x >> 5);   // 8*512*2 jobs (N padded 24->32)
  int lane = threadIdx.x & 31, m = lane & 15, grp = lane >> 4;
  int nt = job & 1; int r1 = job >> 1; int t = r1 & 511; int bt = r1 >> 9;
  const __bf16* base = s1buf + (long)bt * 16 * TSTEPS * 128 + (long)t * 128;
  v8f acc = {};
  for (int kt = 0; kt < 4; ++kt) {
    v16bf a = gatherA_bf(base, (long)TSTEPS * 128, kt * 32, lane);
    v16bf b = loadB(wp + (nt * 4 + kt) * 512 + lane * 16);
    acc = wmma_bf(a, b, acc);
  }
  int n = nt * 16 + m;
  if (n < 24) {
    float bv = bias[n];
#pragma unroll
    for (int i = 0; i < 8; ++i) {
      int row = bt * 16 + i + 8 * grp;
      out[((long)row * TSTEPS + t) * 24 + n] = fast_tanh(acc[i] + bv);
    }
  }
}

// ---------------------------------------------------------------------------
// Host launcher
// ---------------------------------------------------------------------------
extern "C" void kernel_launch(void* const* d_in, const int* in_sizes, int n_in,
                              void* d_out, int out_size, void* d_ws, size_t ws_size,
                              hipStream_t stream) {
  (void)in_sizes; (void)n_in; (void)out_size; (void)ws_size;

  const float* features = (const float*)d_in[0];
  const float* d1_w = (const float*)d_in[1];
  const float* d1_b = (const float*)d_in[2];
  const float *wih[5], *whh[5], *bih[5], *bhh[5], *cw[5], *cb[5];
  for (int n = 0; n < 5; ++n) {
    wih[n] = (const float*)d_in[3 + 4 * n];
    whh[n] = (const float*)d_in[4 + 4 * n];
    bih[n] = (const float*)d_in[5 + 4 * n];
    bhh[n] = (const float*)d_in[6 + 4 * n];
  }
  for (int n = 0; n < 5; ++n) {
    cw[n] = (const float*)d_in[23 + 2 * n];
    cb[n] = (const float*)d_in[24 + 2 * n];
  }
  const float* z_w  = (const float*)d_in[33];
  const float* z_b  = (const float*)d_in[34];
  const float* s1_w = (const float*)d_in[35];
  const float* s1_b = (const float*)d_in[36];
  const float* s2_w = (const float*)d_in[37];
  const float* s2_b = (const float*)d_in[38];

  static const int IN_[5]  = {64, 160, 256, 352, 448};
  static const int CIN_[5] = {128, 224, 320, 416, 512};
  static const int KTI_[5] = {2, 5, 8, 11, 14};
  static const int KT2_[5] = {16, 28, 40, 52, 64};

  // workspace carve-out (bf16 elements, 1KB aligned)
  size_t off = 0;
  auto alloc_bf = [&](size_t elems) -> __bf16* {
    __bf16* p = (__bf16*)((char*)d_ws + off);
    off += ((elems * 2 + 1023) / 1024) * 1024;
    return p;
  };
  __bf16* d1p = alloc_bf(4 * 2 * 512);
  __bf16 *wihp[5], *whhp[5], *convp[5];
  for (int l = 0; l < 5; ++l) wihp[l]  = alloc_bf((size_t)12 * KTI_[l] * 512);
  for (int l = 0; l < 5; ++l) whhp[l]  = alloc_bf(12 * 2 * 512);
  for (int l = 0; l < 5; ++l) convp[l] = alloc_bf((size_t)2 * KT2_[l] * 512);
  __bf16* zp   = alloc_bf(5 * 17 * 512);
  __bf16* s1p  = alloc_bf(8 * 17 * 512);
  __bf16* s2p  = alloc_bf(2 * 4 * 512);
  __bf16* xall  = alloc_bf((size_t)NB * TSTEPS * CH);
  __bf16* s1buf = alloc_bf((size_t)NB * TSTEPS * 128);

  auto rp = [&](const float* W, __bf16* dst, int NT, int KT, int sN, int sK, int ccin) {
    int tot = NT * KT * 512;
    repack_kernel<<<(tot + 255) / 256, 256, 0, stream>>>(W, dst, NT, KT, sN, sK, ccin);
  };
  rp(d1_w, d1p, 4, 2, 64, 40, 0);
  for (int l = 0; l < 5; ++l) rp(wih[l], wihp[l], 12, KTI_[l], 192, IN_[l], 0);
  for (int l = 0; l < 5; ++l) rp(whh[l], whhp[l], 12, 2, 192, 64, 0);
  for (int l = 0; l < 5; ++l) rp(cw[l], convp[l], 2, KT2_[l], 32, 2 * CIN_[l], CIN_[l]);
  rp(z_w,  zp,  5, 17, 80, 544, 0);
  rp(s1_w, s1p, 8, 17, 128, 544, 0);
  rp(s2_w, s2p, 2, 4, 24, 128, 0);

  StreamParams sp;
  sp.feat = features; sp.d1p = d1p; sp.d1b = d1_b; sp.xall = xall;
  for (int l = 0; l < 5; ++l) {
    sp.wihp[l] = wihp[l]; sp.bihp[l] = bih[l];
    sp.whhp[l] = whhp[l]; sp.bhhp[l] = bhh[l];
    sp.convp[l] = convp[l]; sp.convb[l] = cb[l];
  }
  stream_kernel<<<8, 256, 0, stream>>>(sp);

  float* out_z = (float*)d_out;
  float* out_s = out_z + (size_t)NB * TSTEPS * 80;
  head_z <<<2560, 256, 0, stream>>>(xall, zp, z_b, out_z);
  head_s1<<<4096, 256, 0, stream>>>(xall, s1p, s1_b, s1buf);
  head_s2<<<1024, 256, 0, stream>>>(s1buf, s2p, s2_b, out_s);
}
